// LongformerAttention_19834158973061
// MI455X (gfx1250) — matrix operand order repaired and checked by reading, not compiled, not run on hardware
//
#include <hip/hip_runtime.h>

// Longformer attention: B=2, H=16, N=2048, D=64, window=256, global tokens {0,1,2,3}.
// prep: Q * (1/sqrt(D)) * log2(e) -> f16, K -> f16 (row-major), V -> f16 transposed Vt[d][n].
// main: banded flash attention in log2 domain (bare v_exp_f32), one wave per 32 queries
// (two 16-row tiles sharing all K/V loads). Computes S^T = K*Q^T so P lands directly in
// the f16 A-operand register layout (no cross-lane transposes).

#define BH_TOTAL 32            // B*H
#define NSEQ     2048
#define DHEAD    64
#define WIN      256
#define NCHUNKS  (NSEQ / 32)   // 64
#define QPAIRS   (NSEQ / 32)   // 64 query-pairs (32 queries each) per (b,h)

typedef _Float16 v8h  __attribute__((ext_vector_type(8)));
typedef _Float16 v16h __attribute__((ext_vector_type(16)));
typedef float    v8f  __attribute__((ext_vector_type(8)));

// Raw v_exp_f32 (no denormal-range fixup sequence) -- softmax never needs denormal exps.
#if __has_builtin(__builtin_amdgcn_exp2f)
#define FAST_EXP2(x) __builtin_amdgcn_exp2f(x)
#else
#define FAST_EXP2(x) __exp2f(x)
#endif

static __device__ inline v16h cat8(v8h lo, v8h hi) {
  return __builtin_shufflevector(lo, hi, 0,1,2,3,4,5,6,7,8,9,10,11,12,13,14,15);
}

static __device__ inline v8f wmma_f16(v16h a, v16h b, v8f c) {
  return __builtin_amdgcn_wmma_f32_16x16x32_f16(false, a, false, b, (short)0, c, false, false);
}

// Broadcast from a compile-time lane via v_readlane (SGPR path, no LDS traffic).
static __device__ inline float bcast_lane(float x, int l) {
  return __builtin_bit_cast(float, __builtin_amdgcn_readlane(__builtin_bit_cast(int, x), l));
}

// lane i <-> lane i^16 exchange via v_permlanex16 (pure VALU).
static __device__ inline float xor16(float x) {
  int i = __builtin_bit_cast(int, x);
  int r = __builtin_amdgcn_permlanex16(i, i, 0x76543210, 0xfedcba98, false, false);
  return __builtin_bit_cast(float, r);
}

// A-operand (16x32 f16), rows = M, contraction contiguous in memory.
static __device__ inline v16h load_a16x32(const _Float16* p, int ld, int rbase, int cbase, int lane) {
  const int r  = rbase + (lane & 15);
  const int hi = lane >> 4;
  const _Float16* row = p + (size_t)r * ld + cbase;
  v8h lo8 = *(const v8h*)(row + hi * 8);
  v8h hi8 = *(const v8h*)(row + 16 + hi * 8);
  return cat8(lo8, hi8);
}

// B-operand (32x16 f16) where B[K,N] = Src[N,K], Src row-major, contraction contiguous.
static __device__ inline v16h load_b32x16(const _Float16* p, int ld, int nbase, int kbase, int lane) {
  const int n  = nbase + (lane & 15);
  const int hi = lane >> 4;
  const _Float16* row = p + (size_t)n * ld + kbase;
  v8h lo8 = *(const v8h*)(row + hi * 8);
  v8h hi8 = *(const v8h*)(row + 16 + hi * 8);
  return cat8(lo8, hi8);
}

// ---------------- prep: f32 -> f16 (Q pre-scaled by log2e/sqrt(D)), LDS-transposed V ------
__global__ __launch_bounds__(256) void lf_prep_kernel(const float* __restrict__ Q,
                                                      const float* __restrict__ K,
                                                      const float* __restrict__ V,
                                                      _Float16* __restrict__ Qh,
                                                      _Float16* __restrict__ Kh,
                                                      _Float16* __restrict__ Vt) {
  __shared__ _Float16 tile[64][66];
  const int bh = blockIdx.x >> 5;
  const int nt = blockIdx.x & 31;
  const size_t base = ((size_t)bh * NSEQ + (size_t)nt * 64) * DHEAD;
  const int tid = threadIdx.x;
  const float qscale = 0.125f * 1.44269504088896340736f;   // 1/sqrt(64) * log2(e)
  #pragma unroll
  for (int i = tid; i < 64 * DHEAD; i += 256) {
    Qh[base + i] = (_Float16)(Q[base + i] * qscale);
    Kh[base + i] = (_Float16)K[base + i];
    tile[i >> 6][i & 63] = (_Float16)V[base + i];
  }
  __syncthreads();
  #pragma unroll
  for (int i = tid; i < 64 * DHEAD; i += 256) {
    const int d = i >> 6, n = i & 63;
    Vt[((size_t)bh * DHEAD + d) * NSEQ + (size_t)nt * 64 + n] = tile[n][d];
  }
}

// ---------------- main: 1 wave = 32 queries (two 16-row tiles sharing K/V loads) ----------
__global__ __launch_bounds__(256) void lf_attn_kernel(const _Float16* __restrict__ Qh,
                                                      const _Float16* __restrict__ Kh,
                                                      const _Float16* __restrict__ Vt,
                                                      float* __restrict__ out) {
  const int lane = threadIdx.x & 31;
  const int wave = __builtin_amdgcn_readfirstlane((int)(threadIdx.x >> 5));  // wave-uniform
  const int pair = blockIdx.x * 8 + wave;
  const int bh   = pair >> 6;
  const int qp   = pair & (QPAIRS - 1);
  const int qbase = qp << 5;
  const int qloc = lane & 15;
  const int hi   = lane >> 4;

  const _Float16* Qp = Qh + (size_t)bh * NSEQ * DHEAD;
  const _Float16* Kp = Kh + (size_t)bh * NSEQ * DHEAD;
  const _Float16* Vp = Vt + (size_t)bh * DHEAD * NSEQ;
  float*          Op = out + (size_t)bh * NSEQ * DHEAD;

  const v16h qbA0 = load_b32x16(Qp, DHEAD, qbase,      0,  lane);
  const v16h qbA1 = load_b32x16(Qp, DHEAD, qbase,      32, lane);
  const v16h qbB0 = load_b32x16(Qp, DHEAD, qbase + 16, 0,  lane);
  const v16h qbB1 = load_b32x16(Qp, DHEAD, qbase + 16, 32, lane);

  v8f aA0 = {}, aA1 = {}, aA2 = {}, aA3 = {};
  v8f aB0 = {}, aB1 = {}, aB2 = {}, aB3 = {};
  float mA = -3.0e38f, lA = 0.0f;
  float mB = -3.0e38f, lB = 0.0f;

  const int clo = (qbase > WIN) ? ((qbase - WIN) >> 5) : 0;
  int chi = (qbase + 31 + WIN) >> 5;
  if (qp == 0) chi = NCHUNKS - 1;                  // pair 0 holds global queries 0..3
  if (chi > NCHUNKS - 1) chi = NCHUNKS - 1;

  // Scores are already in the log2 domain (scale folded into Qh) -> bare v_exp_f32.
  auto softmax_pack = [&](const v8f& st0, const v8f& st1, int kb, int qb, bool needMask,
                          float& m, float& l, float& alpha) -> v16h {
    const int q = qb + qloc;
    float s[16];
    #pragma unroll
    for (int v = 0; v < 8; ++v) { s[v] = st0[v]; s[8 + v] = st1[v]; }
    if (needMask) {
      const bool qg = (q < 4);                        // global query row
      const bool g0 = (kb == 0) && (hi == 0);         // k<4 only possible here
      const int dqh = q - kb + WIN - hi * 8;          // window test base
      #pragma unroll
      for (int v = 0; v < 16; ++v) {
        const int tbl = (v & 7) + ((v & 8) << 1);     // compile-time element offset
        bool ok = (unsigned)(dqh - tbl) <= (unsigned)(2 * WIN);
        ok = ok || qg;
        if (tbl < 4) ok = ok || g0;                   // global keys 0..3
        s[v] = ok ? s[v] : -3.0e38f;
      }
    }
    float cmax = s[0];
    #pragma unroll
    for (int v = 1; v < 16; ++v) cmax = fmaxf(cmax, s[v]);
    cmax = fmaxf(cmax, xor16(cmax));
    const float mnew = fmaxf(m, cmax);
    alpha = FAST_EXP2(m - mnew);
    v16h pa;
    float csum = 0.0f;
    #pragma unroll
    for (int v = 0; v < 16; ++v) {
      const float pv = FAST_EXP2(s[v] - mnew);
      csum += pv;
      pa[v] = (_Float16)pv;
    }
    csum += xor16(csum);
    l = l * alpha + csum;
    m = mnew;
    return pa;
  };

  auto do_chunk = [&](int kb, bool needMask) {
    const v16h ka00 = load_a16x32(Kp, DHEAD, kb,      0,  lane);
    const v16h ka01 = load_a16x32(Kp, DHEAD, kb,      32, lane);
    const v16h ka10 = load_a16x32(Kp, DHEAD, kb + 16, 0,  lane);
    const v16h ka11 = load_a16x32(Kp, DHEAD, kb + 16, 32, lane);

    v8f st0 = {}, st1 = {};
    st0 = wmma_f16(ka00, qbA0, st0); st0 = wmma_f16(ka01, qbA1, st0);
    st1 = wmma_f16(ka10, qbA0, st1); st1 = wmma_f16(ka11, qbA1, st1);
    float alphaA;
    const v16h paA = softmax_pack(st0, st1, kb, qbase, needMask, mA, lA, alphaA);

    v8f su0 = {}, su1 = {};
    su0 = wmma_f16(ka00, qbB0, su0); su0 = wmma_f16(ka01, qbB1, su0);
    su1 = wmma_f16(ka10, qbB0, su1); su1 = wmma_f16(ka11, qbB1, su1);
    float alphaB;
    const v16h paB = softmax_pack(su0, su1, kb, qbase + 16, needMask, mB, lB, alphaB);

    #pragma unroll
    for (int v = 0; v < 8; ++v) {
      const float a0 = bcast_lane(alphaA, v), a1 = bcast_lane(alphaA, v + 8);
      const float av = hi ? a1 : a0;
      aA0[v] *= av; aA1[v] *= av; aA2[v] *= av; aA3[v] *= av;
      const float b0 = bcast_lane(alphaB, v), b1 = bcast_lane(alphaB, v + 8);
      const float bv = hi ? b1 : b0;
      aB0[v] *= bv; aB1[v] *= bv; aB2[v] *= bv; aB3[v] *= bv;
    }

    v16h vb;
    vb = load_b32x16(Vp, NSEQ, 0,  kb, lane); aA0 = wmma_f16(paA, vb, aA0); aB0 = wmma_f16(paB, vb, aB0);
    vb = load_b32x16(Vp, NSEQ, 16, kb, lane); aA1 = wmma_f16(paA, vb, aA1); aB1 = wmma_f16(paB, vb, aB1);
    vb = load_b32x16(Vp, NSEQ, 32, kb, lane); aA2 = wmma_f16(paA, vb, aA2); aB2 = wmma_f16(paB, vb, aB2);
    vb = load_b32x16(Vp, NSEQ, 48, kb, lane); aA3 = wmma_f16(paA, vb, aA3); aB3 = wmma_f16(paB, vb, aB3);
  };

  if (clo > 0) do_chunk(0, true);                  // global-key chunk (k>=4 masked off)
  for (int c = clo; c <= chi; ++c) {
    const int kb = c << 5;
    if (c < chi) {
      __builtin_prefetch(Kp + (size_t)(kb + 32 + lane) * DHEAD, 0, 0);
    }
    const bool needMask = (kb < qbase - 225) || (kb > qbase + 225);
    do_chunk(kb, needMask);
  }

  const float rlA = 1.0f / lA;
  const float rlB = 1.0f / lB;
  #pragma unroll
  for (int v = 0; v < 8; ++v) {
    const float ra = hi ? bcast_lane(rlA, v + 8) : bcast_lane(rlA, v);
    float* rowA = Op + (size_t)(qbase + v + hi * 8) * DHEAD + qloc;
    rowA[0]  = aA0[v] * ra;
    rowA[16] = aA1[v] * ra;
    rowA[32] = aA2[v] * ra;
    rowA[48] = aA3[v] * ra;
    const float rb = hi ? bcast_lane(rlB, v + 8) : bcast_lane(rlB, v);
    float* rowB = Op + (size_t)(qbase + 16 + v + hi * 8) * DHEAD + qloc;
    rowB[0]  = aB0[v] * rb;
    rowB[16] = aB1[v] * rb;
    rowB[32] = aB2[v] * rb;
    rowB[48] = aB3[v] * rb;
  }
}

extern "C" void kernel_launch(void* const* d_in, const int* in_sizes, int n_in,
                              void* d_out, int out_size, void* d_ws, size_t ws_size,
                              hipStream_t stream) {
  (void)in_sizes; (void)n_in; (void)out_size; (void)ws_size;
  const float* Q = (const float*)d_in[0];
  const float* K = (const float*)d_in[1];
  const float* V = (const float*)d_in[2];
  float* out = (float*)d_out;

  const size_t plane = (size_t)BH_TOTAL * NSEQ * DHEAD;
  _Float16* Qh = (_Float16*)d_ws;
  _Float16* Kh = Qh + plane;
  _Float16* Vt = Kh + plane;

  lf_prep_kernel<<<BH_TOTAL * (NSEQ / 64), 256, 0, stream>>>(Q, K, V, Qh, Kh, Vt);
  lf_attn_kernel<<<(BH_TOTAL * QPAIRS) / 8, 256, 0, stream>>>(Qh, Kh, Vt, out);
}
